// Zamba2MambaDecoderLayer_90082644066663
// MI455X (gfx1250) — matrix operand order repaired and checked by reading, not compiled, not run on hardware
//
#include <hip/hip_runtime.h>
#include <hip/hip_bf16.h>

#define DEV __device__ __forceinline__

// ---- problem dims (fixed by reference) ----
static constexpr int B_   = 2;
static constexpr int L_   = 2048;
static constexpr int DM   = 2048;
static constexpr int DI   = 4096;              // d_inner
static constexpr int NST  = 64;                // d_state
static constexpr int KC   = 4;                 // d_conv
static constexpr int P_   = 64;                // headdim
static constexpr int H_   = 64;                // heads
static constexpr int CONV = DI + 2 * NST;      // 4224
static constexpr int PROJ = 2 * DI + 2 * NST + H_; // 8384
static constexpr int ROWS = B_ * L_;           // 4096
static constexpr float EPS = 1e-6f;

typedef __attribute__((ext_vector_type(16))) __bf16 v16bf;
typedef __attribute__((ext_vector_type(8)))  float  v8f;

DEV unsigned short f2bf(float f) {
  unsigned int u = __float_as_uint(f);
  u += 0x7FFFu + ((u >> 16) & 1u);   // round-to-nearest-even
  return (unsigned short)(u >> 16);
}

DEV float siluf(float x) { return x / (1.f + expf(-x)); }

// ---- CDNA5 async global->LDS DMA (ASYNCcnt-tracked) ----
// LDS offset = low 32 bits of the generic (flat) shared-memory address.
DEV unsigned int lds_off(const void* p) { return (unsigned int)(uintptr_t)p; }

// INST_OFFSET applies to both the LDS and global address (ISA 08_async_tensor §4.4).
DEV void async_cp64(unsigned int loff, const void* gptr) {   // 64B per lane
  unsigned long long g = (unsigned long long)(uintptr_t)gptr;
  asm volatile("global_load_async_to_lds_b128 %0, %1, off\n\t"
               "global_load_async_to_lds_b128 %0, %1, off offset:16\n\t"
               "global_load_async_to_lds_b128 %0, %1, off offset:32\n\t"
               "global_load_async_to_lds_b128 %0, %1, off offset:48"
               :: "v"(loff), "v"(g) : "memory");
}
DEV void async_cp16(unsigned int loff, const void* gptr) {   // 16B per lane
  unsigned long long g = (unsigned long long)(uintptr_t)gptr;
  asm volatile("global_load_async_to_lds_b128 %0, %1, off"
               :: "v"(loff), "v"(g) : "memory");
}
DEV void async_cp4(unsigned int loff, const void* gptr) {    // 4B per lane
  unsigned long long g = (unsigned long long)(uintptr_t)gptr;
  asm volatile("global_load_async_to_lds_b32 %0, %1, off"
               :: "v"(loff), "v"(g) : "memory");
}
DEV void wait_async0() { asm volatile("s_wait_asynccnt 0x0" ::: "memory"); }

DEV float block_reduce_sum256(float v, float* sbuf) {
  #pragma unroll
  for (int o = 16; o; o >>= 1) v += __shfl_xor(v, o, 32);
  int w = threadIdx.x >> 5;
  if ((threadIdx.x & 31) == 0) sbuf[w] = v;
  __syncthreads();
  if (threadIdx.x < 32) {
    float t = (threadIdx.x < 8) ? sbuf[threadIdx.x] : 0.f;
    #pragma unroll
    for (int o = 4; o; o >>= 1) t += __shfl_xor(t, o, 32);
    if (threadIdx.x == 0) sbuf[0] = t;
  }
  __syncthreads();
  return sbuf[0];
}

// ---------------- f32 -> bf16 weight cast ----------------
__global__ __launch_bounds__(256) void cast_bf16_kernel(const float* __restrict__ in,
                                                        unsigned short* __restrict__ out,
                                                        int n) {
  int i = (blockIdx.x * 256 + threadIdx.x) * 4;
  if (i >= n) return;
  float4 v = *(const float4*)(in + i);
  unsigned short r[4] = { f2bf(v.x), f2bf(v.y), f2bf(v.z), f2bf(v.w) };
  *(uint2*)(out + i) = *(uint2*)r;
}

// ---------------- input RMSNorm -> bf16 ----------------
__global__ __launch_bounds__(256) void rmsnorm_cast_kernel(const float* __restrict__ x,
                                                           const float* __restrict__ w,
                                                           unsigned short* __restrict__ out,
                                                           int ncols) {
  __shared__ float sred[8];
  int row = blockIdx.x;
  const float* xr = x + (size_t)row * ncols;
  float ss = 0.f;
  for (int c = threadIdx.x; c < ncols; c += 256) { float v = xr[c]; ss += v * v; }
  float tot = block_reduce_sum256(ss, sred);
  float scale = rsqrtf(tot / (float)ncols + EPS);
  unsigned short* orow = out + (size_t)row * ncols;
  for (int c = threadIdx.x; c < ncols; c += 256) orow[c] = f2bf(xr[c] * scale * w[c]);
}

// ---------------- WMMA GEMM: C[M,N] = A[M,K] * B[N,K]^T (+residual) ----------------
// bf16 inputs, f32 accumulate. BLK 128x128, K-slab 64 per LDS buffer. 256 thr =
// 8 wave32 (4x2 wave grid), each wave 2x4 fragments of 16x16. Double-buffered LDS
// (2 x 32KB) filled by async-to-LDS DMA; one barrier per 16 WMMAs.
__global__ __launch_bounds__(256) void wmma_gemm_kernel(const unsigned short* __restrict__ A,
                                                        const unsigned short* __restrict__ Bw,
                                                        const float* __restrict__ Res,
                                                        float* __restrict__ C,
                                                        int M, int N, int Kt) {
  __shared__ __align__(16) unsigned short As[2][128 * 64];
  __shared__ __align__(16) unsigned short Bs[2][128 * 64];
  const int tid  = threadIdx.x;
  const int lane = tid & 31;
  const int wid  = tid >> 5;
  const int wm   = wid & 3;      // 32-row slab
  const int wn   = wid >> 2;     // 64-col slab
  const int m0 = blockIdx.y * 128;
  const int n0 = blockIdx.x * 128;

  v8f acc[2][4];
  #pragma unroll
  for (int i = 0; i < 2; ++i)
    #pragma unroll
    for (int j = 0; j < 4; ++j)
      #pragma unroll
      for (int e = 0; e < 8; ++e) acc[i][j][e] = 0.f;

  const int ldRow = tid >> 1;          // 0..127
  const int ldK   = (tid & 1) * 32;    // 0 or 32 (elements)
  const unsigned short* Aptr = A + (size_t)(m0 + ldRow) * Kt + ldK;
  int gnc = n0 + ldRow;                // clamp OOB rows of B: clamped garbage only
  if (gnc > N - 1) gnc = N - 1;        // feeds columns >= N, which are never stored
  const unsigned short* Bptr = Bw + (size_t)gnc * Kt + ldK;

  unsigned int lA[2], lB[2];
  lA[0] = lds_off(&As[0][ldRow * 64 + ldK]);
  lA[1] = lds_off(&As[1][ldRow * 64 + ldK]);
  lB[0] = lds_off(&Bs[0][ldRow * 64 + ldK]);
  lB[1] = lds_off(&Bs[1][ldRow * 64 + ldK]);

  // fragment lane addressing (ISA 7.12.2 16-bit layouts, wave32)
  const int half = lane >> 4;          // 0: lanes 0-15, 1: lanes 16-31
  const int l15  = lane & 15;
  const int a_kb = half * 8;           // A: K base {0,8}, plus +16 block
  const int b_kb = half * 8;           // B: K base {0,8}, plus +16 block

  // prologue: stage K-slab 0
  async_cp64(lA[0], Aptr);
  async_cp64(lB[0], Bptr);
  wait_async0();
  __syncthreads();

  const int nk = Kt >> 6;              // 64-wide K slabs
  for (int kb = 0; kb < nk; ++kb) {
    const int cur = kb & 1;
    if (kb + 1 < nk) {                 // stage next slab while computing this one
      async_cp64(lA[cur ^ 1], Aptr + (kb + 1) * 64);
      async_cp64(lB[cur ^ 1], Bptr + (kb + 1) * 64);
    }

    union Frag { v16bf v; uint4 q[2]; };
    #pragma unroll
    for (int ks = 0; ks < 2; ++ks) {   // two K=32 WMMA steps per slab
      const int kof = ks * 32;
      Frag af[2], bf[4];
      #pragma unroll
      for (int im = 0; im < 2; ++im) {
        int r = wm * 32 + im * 16 + l15;
        af[im].q[0] = *(const uint4*)(&As[cur][r * 64 + kof + a_kb]);      // K kb..kb+7
        af[im].q[1] = *(const uint4*)(&As[cur][r * 64 + kof + a_kb + 16]); // K kb+16..kb+23
      }
      #pragma unroll
      for (int jn = 0; jn < 4; ++jn) {
        int cL = wn * 64 + jn * 16 + l15;
        bf[jn].q[0] = *(const uint4*)(&Bs[cur][cL * 64 + kof + b_kb]);      // K kb..kb+7
        bf[jn].q[1] = *(const uint4*)(&Bs[cur][cL * 64 + kof + b_kb + 16]); // K kb+16..kb+23
      }
      #pragma unroll
      for (int im = 0; im < 2; ++im)
        #pragma unroll
        for (int jn = 0; jn < 4; ++jn)
          acc[im][jn] = __builtin_amdgcn_wmma_f32_16x16x32_bf16(
              false, af[im].v, false, bf[jn].v, (short)0, acc[im][jn], false, false);
    }

    wait_async0();      // next-slab DMA complete (own waves; barrier covers others)
    __syncthreads();
  }

  // epilogue: C layout (f32 16x16): lane l -> N = l&15, VGPR e -> M = e + 8*(l>>4)
  const int rbase = m0 + wm * 32 + half * 8;
  const int cbase = n0 + wn * 64 + l15;
  #pragma unroll
  for (int im = 0; im < 2; ++im) {
    #pragma unroll
    for (int jn = 0; jn < 4; ++jn) {
      int col = cbase + jn * 16;
      if (col < N) {
        #pragma unroll
        for (int e = 0; e < 8; ++e) {
          int rowg = rbase + im * 16 + e;
          size_t off = (size_t)rowg * N + col;
          float v = acc[im][jn][e];
          if (Res) v += Res[off];
          C[off] = v;
        }
      }
    }
  }
}

// ---------------- causal depthwise conv1d (K=4) + SiLU ----------------
__global__ __launch_bounds__(256) void conv_silu_kernel(const float* __restrict__ zx,
                                                        const float* __restrict__ cw,
                                                        const float* __restrict__ cb,
                                                        float* __restrict__ xc) {
  size_t idx = (size_t)blockIdx.x * 256 + threadIdx.x;
  if (idx >= (size_t)ROWS * CONV) return;
  int c = (int)(idx % CONV);
  size_t row = idx / CONV;          // b*L + l
  int l = (int)(row % L_);
  float acc = cb[c];
  #pragma unroll
  for (int k = 0; k < KC; ++k) {
    int t = l + k - (KC - 1);
    if (t >= 0) acc += zx[(row + (size_t)(t - l)) * PROJ + DI + c] * cw[k * CONV + c];
  }
  xc[idx] = siluf(acc);
}

// ---------------- softplus(dt)+dA precompute ----------------
__global__ __launch_bounds__(256) void dt_kernel(const float* __restrict__ zx,
                                                 const float* __restrict__ dt_bias,
                                                 const float* __restrict__ A_log,
                                                 float* __restrict__ dts,
                                                 float* __restrict__ dAs) {
  int idx = blockIdx.x * 256 + threadIdx.x;
  if (idx >= ROWS * H_) return;
  int h = idx & (H_ - 1);
  int row = idx >> 6;
  float x = zx[(size_t)row * PROJ + DI + CONV + h] + dt_bias[h];
  float dt = (x > 20.f) ? x : log1pf(expf(x));
  dts[idx] = dt;
  dAs[idx] = expf(-expf(A_log[h]) * dt);
}

// ---------------- SSM sequential scan ----------------
// 1 WG per (b, h, p-half): 256 WGs. Thread t: p_local = t>>3, n-range = (t&7)*8..+8.
// 8 f32 states/thread in registers. Chunks of 16 timesteps double-buffered through
// LDS via async-to-LDS DMA so chunk c+1 streams in while chunk c is recursed.
static constexpr int TCH = 16;
struct ScanBuf {
  float B[TCH][NST];   // 4096 B
  float C[TCH][NST];   // 4096 B
  float X[TCH][32];    // 2048 B
  float dA[TCH];       // 64 B
  float dt[TCH];       // 64 B
};
__global__ __launch_bounds__(256) void scan_kernel(const float* __restrict__ xc,
                                                   const float* __restrict__ dts,
                                                   const float* __restrict__ dAs,
                                                   const float* __restrict__ Dp,
                                                   float* __restrict__ y) {
  __shared__ __align__(16) ScanBuf sb[2];
  const int wg = blockIdx.x;
  const int b = wg >> 7;                 // / (H_*2)
  const int rem = wg & 127;
  const int h = rem >> 1;
  const int pblk = rem & 1;
  const int tid = threadIdx.x;
  const int pl = tid >> 3;               // 0..31
  const int p  = pblk * 32 + pl;
  const int nsub = tid & 7;
  const int nb = nsub * 8;
  float st[8];
  #pragma unroll
  for (int j = 0; j < 8; ++j) st[j] = 0.f;
  const float Dh = Dp[h];
  const size_t rowbase = (size_t)b * L_;

  // per-thread staging roles (16B granules)
  const int tB = tid >> 4, qB = tid & 15;           // B/C rows: 16 rows x 16 granules
  const int tX = tid >> 3, qX = tid & 7;            // X rows: 16 rows x 8 granules (tid<128)
  const size_t offB = (size_t)tB * CONV + DI + qB * 4;
  const size_t offC = offB + NST;
  const size_t offX = (size_t)tX * CONV + h * P_ + pblk * 32 + qX * 4;
  unsigned int lBo[2], lCo[2], lXo[2], lSo[2];
  #pragma unroll
  for (int u = 0; u < 2; ++u) {
    lBo[u] = lds_off(&sb[u].B[tB][qB * 4]);
    lCo[u] = lds_off(&sb[u].C[tB][qB * 4]);
    lXo[u] = lds_off(&sb[u].X[tX & 15][qX * 4]);
    int t16 = tid & 15;
    lSo[u] = (tid >= 128 && tid < 144) ? lds_off(&sb[u].dA[t16])
                                       : lds_off(&sb[u].dt[t16]);
  }

  auto stage = [&](int buf, int c) {
    const float* base = xc + (rowbase + (size_t)c * TCH) * CONV;
    async_cp16(lBo[buf], base + offB);
    async_cp16(lCo[buf], base + offC);
    if (tid < 128) {
      async_cp16(lXo[buf], base + offX);
    } else if (tid < 144) {
      async_cp4(lSo[buf], dAs + (rowbase + (size_t)c * TCH + (tid - 128)) * H_ + h);
    } else if (tid < 160) {
      async_cp4(lSo[buf], dts + (rowbase + (size_t)c * TCH + (tid - 144)) * H_ + h);
    }
  };

  stage(0, 0);
  wait_async0();
  __syncthreads();

  const int nchunks = L_ / TCH;
  for (int c = 0; c < nchunks; ++c) {
    const int cur = c & 1;
    if (c + 1 < nchunks) stage(cur ^ 1, c + 1);
    const size_t l0 = (size_t)c * TCH;
    #pragma unroll 4
    for (int t = 0; t < TCH; ++t) {
      float dAv = sb[cur].dA[t], dtv = sb[cur].dt[t];
      float xp = sb[cur].X[t][pl];
      float dtx = dtv * xp;
      float ys = 0.f;
      #pragma unroll
      for (int j = 0; j < 8; ++j) {
        float s = fmaf(dAv, st[j], dtx * sb[cur].B[t][nb + j]);
        st[j] = s;
        ys = fmaf(s, sb[cur].C[t][nb + j], ys);
      }
      ys += __shfl_xor(ys, 1, 32);
      ys += __shfl_xor(ys, 2, 32);
      ys += __shfl_xor(ys, 4, 32);
      if (nsub == 0)
        y[(rowbase + l0 + t) * DI + h * P_ + p] = ys + Dh * xp;
    }
    wait_async0();
    __syncthreads();
  }
}

// ---------------- gated RMSNorm -> bf16 ----------------
__global__ __launch_bounds__(256) void gated_norm_kernel(const float* __restrict__ y,
                                                         const float* __restrict__ zx,
                                                         const float* __restrict__ w,
                                                         unsigned short* __restrict__ out) {
  __shared__ float sred[8];
  int row = blockIdx.x;
  const float* yr = y  + (size_t)row * DI;
  const float* zr = zx + (size_t)row * PROJ;   // z = first DI columns
  float g[16];
  float ss = 0.f;
  #pragma unroll
  for (int i = 0; i < 16; ++i) {
    int c = threadIdx.x + i * 256;
    float z = zr[c];
    float gv = yr[c] * siluf(z);
    g[i] = gv;
    ss += gv * gv;
  }
  float tot = block_reduce_sum256(ss, sred);
  float scale = rsqrtf(tot / (float)DI + EPS);
  unsigned short* orow = out + (size_t)row * DI;
  #pragma unroll
  for (int i = 0; i < 16; ++i) {
    int c = threadIdx.x + i * 256;
    orow[c] = f2bf(w[c] * g[i] * scale);
  }
}

// ---------------- launch ----------------
extern "C" void kernel_launch(void* const* d_in, const int* in_sizes, int n_in,
                              void* d_out, int out_size, void* d_ws, size_t ws_size,
                              hipStream_t stream) {
  (void)in_sizes; (void)n_in; (void)out_size; (void)ws_size;
  const float* u       = (const float*)d_in[0];
  const float* ln_w    = (const float*)d_in[1];
  const float* in_w    = (const float*)d_in[2];
  const float* conv_w  = (const float*)d_in[3];
  const float* conv_b  = (const float*)d_in[4];
  const float* dt_bias = (const float*)d_in[5];
  const float* A_log   = (const float*)d_in[6];
  const float* Dp      = (const float*)d_in[7];
  const float* norm_w  = (const float*)d_in[8];
  const float* out_w   = (const float*)d_in[9];
  float* out = (float*)d_out;

  char* ws = (char*)d_ws;
  size_t off = 0;
  auto alloc = [&](size_t bytes) {
    char* p = ws + off;
    off = (off + bytes + 255) & ~(size_t)255;
    return p;
  };
  unsigned short* w1b = (unsigned short*)alloc((size_t)PROJ * DM * 2);
  unsigned short* w2b = (unsigned short*)alloc((size_t)DM * DI * 2);
  unsigned short* hsb = (unsigned short*)alloc((size_t)ROWS * DM * 2);
  float*          zx  = (float*)alloc((size_t)ROWS * PROJ * 4);
  float*          xc  = (float*)alloc((size_t)ROWS * CONV * 4);
  float*          dts = (float*)alloc((size_t)ROWS * H_ * 4);
  float*          dAs = (float*)alloc((size_t)ROWS * H_ * 4);
  float*          yv  = (float*)alloc((size_t)ROWS * DI * 4);
  unsigned short* yb  = (unsigned short*)alloc((size_t)ROWS * DI * 2);

  // weight casts to bf16
  {
    int n1 = PROJ * DM, n2 = DM * DI;
    cast_bf16_kernel<<<(n1 / 4 + 255) / 256, 256, 0, stream>>>(in_w,  w1b, n1);
    cast_bf16_kernel<<<(n2 / 4 + 255) / 256, 256, 0, stream>>>(out_w, w2b, n2);
  }
  // input RMSNorm -> bf16
  rmsnorm_cast_kernel<<<ROWS, 256, 0, stream>>>(u, ln_w, hsb, DM);
  // in_proj GEMM (4096 x 8384 x 2048)
  {
    dim3 grid((PROJ + 127) / 128, ROWS / 128);
    wmma_gemm_kernel<<<grid, 256, 0, stream>>>(hsb, w1b, nullptr, zx, ROWS, PROJ, DM);
  }
  // conv + SiLU
  conv_silu_kernel<<<(int)(((size_t)ROWS * CONV + 255) / 256), 256, 0, stream>>>(zx, conv_w, conv_b, xc);
  // softplus(dt)+dA
  dt_kernel<<<(ROWS * H_ + 255) / 256, 256, 0, stream>>>(zx, dt_bias, A_log, dts, dAs);
  // SSM scan
  scan_kernel<<<B_ * H_ * 2, 256, 0, stream>>>(xc, dts, dAs, Dp, yv);
  // gated RMSNorm -> bf16
  gated_norm_kernel<<<ROWS, 256, 0, stream>>>(yv, zx, norm_w, yb);
  // out_proj GEMM + residual (4096 x 2048 x 4096)
  {
    dim3 grid(DM / 128, ROWS / 128);
    wmma_gemm_kernel<<<grid, 256, 0, stream>>>(yb, w2b, u, out, ROWS, DM, DI);
  }
}
